// NISER_4964982194193
// MI455X (gfx1250) — matrix operand order repaired and checked by reading, not compiled, not run on hardware
//
#include <hip/hip_runtime.h>
#include <hip/hip_bf16.h>

typedef __bf16 bf16_t;
typedef __attribute__((ext_vector_type(16))) __bf16 v16bf;
typedef __attribute__((ext_vector_type(8)))  float  v8f;
typedef __attribute__((ext_vector_type(4)))  unsigned su4;
typedef __attribute__((ext_vector_type(8)))  unsigned su8;

#define HDIM 128
#define NNODE 64
#define BATCH 512
#define SEQL  64
#define VOCAB 100000

__device__ inline v8f wmma_bf16(v16bf a, v16bf b, v8f c) {
  return __builtin_amdgcn_wmma_f32_16x16x32_bf16(false, a, false, b, (short)0, c,
                                                 false, false);
}

// ---- A-operand loaders (16x32 bf16 tile, contiguous K pattern) ----
// lanes 0-15 : row m0+lane, K = kb+{0..15}, kb = 32*kc
// lanes 16-31: row m0+lane-16, same with kb += 16
__device__ inline v16bf ldA_bf16(const bf16_t* M, int ld, int m0, int kc, int lane) {
  int m  = m0 + (lane & 15);
  int kb = kc * 32 + ((lane & 16) ? 16 : 0);
  const bf16_t* p = M + m * ld + kb;
  v16bf out;
#pragma unroll
  for (int i = 0; i < 16; ++i) out[i] = p[i];
  return out;
}
__device__ inline v16bf ldA_f32(const float* M, int ld, int m0, int kc, int lane) {
  int m  = m0 + (lane & 15);
  int kb = kc * 32 + ((lane & 16) ? 16 : 0);
  const float* p = M + m * ld + kb;
  v16bf out;
#pragma unroll
  for (int i = 0; i < 16; ++i) out[i] = (bf16_t)p[i];
  return out;
}

// ---- B-operand loaders (32x16 tile): lane n holds column n0+lane%16,
//      K = 32*kc + (lane<16 ? 0 : 16) + {0..15} ----
// B column n = row n of W (computing X @ W^T, W row-major [N][K])
__device__ inline v16bf ldB_rowT_f32(const float* W, int ld, int n0, int kc, int lane) {
  int n  = n0 + (lane & 15);
  int kb = kc * 32 + ((lane & 16) ? 16 : 0);
  const float* p = W + n * ld + kb;
  v16bf out;
#pragma unroll
  for (int i = 0; i < 16; ++i) out[i] = (bf16_t)p[i];
  return out;
}
// plain X @ W (W row-major [K][N]): B[k][n] = W[k*ld + n]
__device__ inline v16bf ldB_col_f32(const float* W, int ld, int n0, int kc, int lane) {
  int n  = n0 + (lane & 15);
  int kb = kc * 32 + ((lane & 16) ? 16 : 0);
  v16bf out;
#pragma unroll
  for (int i = 0; i < 16; ++i) out[i] = (bf16_t)W[(kb + i) * ld + n];
  return out;
}
__device__ inline v16bf ldB_col_bf16(const bf16_t* W, int ld, int n0, int kc, int lane) {
  int n  = n0 + (lane & 15);
  int kb = kc * 32 + ((lane & 16) ? 16 : 0);
  v16bf out;
#pragma unroll
  for (int i = 0; i < 16; ++i) out[i] = W[(kb + i) * ld + n];
  return out;
}

__device__ inline float sigmoidf(float x) { return 1.f / (1.f + expf(-x)); }

// ============================================================================
// Kernel 1: one block per session b.  Full GNN cell + positional attention.
// Emits a_final[b] (bf16, [BATCH][HDIM]) into workspace.
// ============================================================================
__global__ __launch_bounds__(256)
void niser_session_kernel(const int*   __restrict__ items,
                          const float* __restrict__ Amat,
                          const int*   __restrict__ alias_in,
                          const float* __restrict__ emb,
                          const float* __restrict__ pos,
                          const float* __restrict__ linear_pos,
                          const float* __restrict__ w_ih,
                          const float* __restrict__ w_hh,
                          const float* __restrict__ b_ih,
                          const float* __restrict__ b_hh,
                          const float* __restrict__ b_iah,
                          const float* __restrict__ b_oah,
                          const float* __restrict__ W_in,
                          const float* __restrict__ b_in,
                          const float* __restrict__ W_out,
                          const float* __restrict__ b_out,
                          const float* __restrict__ W1,
                          const float* __restrict__ b1,
                          const float* __restrict__ W2,
                          const float* __restrict__ b2,
                          const float* __restrict__ W3,
                          const float* __restrict__ Wt,
                          const float* __restrict__ bt,
                          bf16_t*      __restrict__ a_out) {
  __shared__ float  sHid [NNODE][HDIM];      // 32 KB  hidden (f32)
  __shared__ bf16_t sHidB[NNODE][HDIM];      // 16 KB  hidden (bf16, A operand)
  __shared__ bf16_t sBufB[NNODE][2 * HDIM];  // 32 KB  hv -> (r | i) gates
  __shared__ bf16_t sBufC[NNODE][2 * HDIM];  // 32 KB  inputs -> X2
  __shared__ float  sNorm[NNODE];
  __shared__ float  sHt[HDIM];
  __shared__ float  sQ1[HDIM];
  __shared__ float  sAlpha[NNODE];
  __shared__ float  sAvec[HDIM];
  __shared__ int    sLast;

  const int b    = blockIdx.x;
  const int tid  = threadIdx.x;
  const int wave = tid >> 5;
  const int lane = tid & 31;
  const int* al  = alias_in + b * SEQL;

  // S0: hidden = emb[items[b]]
  for (int idx = tid; idx < NNODE * HDIM; idx += 256) {
    int l = idx >> 7, h = idx & (HDIM - 1);
    int it = items[b * NNODE + l];
    float v = emb[(size_t)it * HDIM + h];
    sHid[l][h]  = v;
    sHidB[l][h] = (bf16_t)v;
  }
  __syncthreads();

  // S1: hv = [hidden@W_in^T + b_in | hidden@W_out^T + b_out] -> sBufB (bf16)
  for (int g = 0; g < 2; ++g) {
    const float* W    = g ? W_out : W_in;
    const float* bias = g ? b_out : b_in;
    for (int task = wave; task < 32; task += 8) {
      int m0 = (task >> 3) * 16, n0 = (task & 7) * 16;
      v8f acc = {0.f, 0.f, 0.f, 0.f, 0.f, 0.f, 0.f, 0.f};
      for (int kc = 0; kc < 4; ++kc)
        acc = wmma_bf16(ldA_bf16(&sHidB[0][0], HDIM, m0, kc, lane),
                        ldB_rowT_f32(W, HDIM, n0, kc, lane), acc);
      int n = n0 + (lane & 15);
      int mb = m0 + ((lane & 16) ? 8 : 0);
      float bn = bias[n];
#pragma unroll
      for (int e = 0; e < 8; ++e)
        sBufB[mb + e][g * HDIM + n] = (bf16_t)(acc[e] + bn);
    }
  }
  __syncthreads();

  // S2: inputs = [A_in@hv_in + b_iah | A_out@hv_out + b_oah] -> sBufC (bf16)
  const float* Ab = Amat + (size_t)b * NNODE * (2 * NNODE);
  for (int g = 0; g < 2; ++g) {
    const float* Ap   = Ab + g * NNODE;       // cols 0..63 or 64..127
    const float* bias = g ? b_oah : b_iah;
    for (int task = wave; task < 32; task += 8) {
      int m0 = (task >> 3) * 16, n0 = (task & 7) * 16;
      v8f acc = {0.f, 0.f, 0.f, 0.f, 0.f, 0.f, 0.f, 0.f};
      for (int kc = 0; kc < 2; ++kc)  // K = 64
        acc = wmma_bf16(ldA_f32(Ap, 2 * NNODE, m0, kc, lane),
                        ldB_col_bf16(&sBufB[0][0], 2 * HDIM, g * HDIM + n0, kc, lane),
                        acc);
      int n = n0 + (lane & 15);
      int mb = m0 + ((lane & 16) ? 8 : 0);
      float bn = bias[n];
#pragma unroll
      for (int e = 0; e < 8; ++e)
        sBufC[mb + e][g * HDIM + n] = (bf16_t)(acc[e] + bn);
    }
  }
  __syncthreads();

  // S3: reset / input gates: sigmoid(inputs@w_ih[p]^T + hid@w_hh[p]^T + biases)
  for (int p = 0; p < 2; ++p) {
    for (int task = wave; task < 32; task += 8) {
      int m0 = (task >> 3) * 16, n0 = (task & 7) * 16;
      v8f acc = {0.f, 0.f, 0.f, 0.f, 0.f, 0.f, 0.f, 0.f};
      for (int kc = 0; kc < 8; ++kc)  // K = 256
        acc = wmma_bf16(ldA_bf16(&sBufC[0][0], 2 * HDIM, m0, kc, lane),
                        ldB_rowT_f32(w_ih + p * HDIM * (2 * HDIM), 2 * HDIM, n0, kc, lane),
                        acc);
      for (int kc = 0; kc < 4; ++kc)  // K = 128
        acc = wmma_bf16(ldA_bf16(&sHidB[0][0], HDIM, m0, kc, lane),
                        ldB_rowT_f32(w_hh + p * HDIM * HDIM, HDIM, n0, kc, lane), acc);
      int n = n0 + (lane & 15);
      int mb = m0 + ((lane & 16) ? 8 : 0);
      float bn = b_ih[p * HDIM + n] + b_hh[p * HDIM + n];
#pragma unroll
      for (int e = 0; e < 8; ++e)
        sBufB[mb + e][p * HDIM + n] = (bf16_t)sigmoidf(acc[e] + bn);
    }
  }
  __syncthreads();

  // S4: newgate + hidden update (in place on sHid)
  for (int task = wave; task < 32; task += 8) {
    int m0 = (task >> 3) * 16, n0 = (task & 7) * 16;
    v8f ai = {0.f, 0.f, 0.f, 0.f, 0.f, 0.f, 0.f, 0.f};
    for (int kc = 0; kc < 8; ++kc)
      ai = wmma_bf16(ldA_bf16(&sBufC[0][0], 2 * HDIM, m0, kc, lane),
                     ldB_rowT_f32(w_ih + (2 * HDIM) * (2 * HDIM), 2 * HDIM, n0, kc, lane),
                     ai);
    v8f ah = {0.f, 0.f, 0.f, 0.f, 0.f, 0.f, 0.f, 0.f};
    for (int kc = 0; kc < 4; ++kc)
      ah = wmma_bf16(ldA_bf16(&sHidB[0][0], HDIM, m0, kc, lane),
                     ldB_rowT_f32(w_hh + (2 * HDIM) * HDIM, HDIM, n0, kc, lane), ah);
    int n = n0 + (lane & 15);
    int mb = m0 + ((lane & 16) ? 8 : 0);
    float bi = b_ih[2 * HDIM + n], bh = b_hh[2 * HDIM + n];
#pragma unroll
    for (int e = 0; e < 8; ++e) {
      int m = mb + e;
      float rg = (float)sBufB[m][n];
      float ig = (float)sBufB[m][HDIM + n];
      float ng = tanhf(ai[e] + bi + rg * (ah[e] + bh));
      float hc = sHid[m][n];
      sHid[m][n] = ng + ig * (hc - ng);
    }
  }
  __syncthreads();

  // S6: X2 = [pos | hidden[alias]] -> sBufC (bf16)
  for (int idx = tid; idx < SEQL * 2 * HDIM; idx += 256) {
    int l = idx >> 8, k = idx & (2 * HDIM - 1);
    float v = (k < HDIM) ? pos[l * HDIM + k] : sHid[al[l]][k - HDIM];
    sBufC[l][k] = (bf16_t)v;
  }
  __syncthreads();

  // S7: seq = tanh(X2 @ linear_pos) -> sHid (f32, overwrite)
  for (int task = wave; task < 32; task += 8) {
    int m0 = (task >> 3) * 16, n0 = (task & 7) * 16;
    v8f acc = {0.f, 0.f, 0.f, 0.f, 0.f, 0.f, 0.f, 0.f};
    for (int kc = 0; kc < 8; ++kc)
      acc = wmma_bf16(ldA_bf16(&sBufC[0][0], 2 * HDIM, m0, kc, lane),
                      ldB_col_f32(linear_pos, HDIM, n0, kc, lane), acc);
    int n = n0 + (lane & 15);
    int mb = m0 + ((lane & 16) ? 8 : 0);
#pragma unroll
    for (int e = 0; e < 8; ++e) sHid[mb + e][n] = tanhf(acc[e]);
  }
  __syncthreads();

  // S8: row-wise L2 normalize, refresh bf16 copy, last index, alpha init
  if (tid < NNODE) {
    float ss = 0.f;
    for (int h = 0; h < HDIM; ++h) { float v = sHid[tid][h]; ss += v * v; }
    sNorm[tid] = rsqrtf(ss);
  }
  if (tid == 0) {
    int cnt = 0;
    for (int l = 0; l < SEQL; ++l) cnt += (al[l] > 0);
    int last = cnt - 1;
    if (last < 0) last += SEQL;   // emulate python -1 wraparound
    sLast = last;
  }
  if (tid < NNODE) sAlpha[tid] = 0.f;
  __syncthreads();
  for (int idx = tid; idx < NNODE * HDIM; idx += 256) {
    int l = idx >> 7, h = idx & (HDIM - 1);
    float v = sHid[l][h] * sNorm[l];
    sHid[l][h]  = v;
    sHidB[l][h] = (bf16_t)v;
  }
  __syncthreads();

  // S9: ht, q1 = ht@W1^T + b1
  if (tid < HDIM) sHt[tid] = sHid[sLast][tid];
  __syncthreads();
  if (tid < HDIM) {
    float q = b1[tid];
    const float* wr = W1 + tid * HDIM;
    for (int k = 0; k < HDIM; ++k) q += sHt[k] * wr[k];
    sQ1[tid] = q;
  }
  __syncthreads();

  // S10: q2 = seq@W2^T + b2;  alpha[l] = sum_h sigmoid(q1+q2)[l,h] * W3[h]
  for (int task = wave; task < 32; task += 8) {
    int m0 = (task >> 3) * 16, n0 = (task & 7) * 16;
    v8f acc = {0.f, 0.f, 0.f, 0.f, 0.f, 0.f, 0.f, 0.f};
    for (int kc = 0; kc < 4; ++kc)
      acc = wmma_bf16(ldA_bf16(&sHidB[0][0], HDIM, m0, kc, lane),
                      ldB_rowT_f32(W2, HDIM, n0, kc, lane), acc);
    int n = n0 + (lane & 15);
    int mb = m0 + ((lane & 16) ? 8 : 0);
    float b2n = b2[n], w3n = W3[n], q1n = sQ1[n];
#pragma unroll
    for (int e = 0; e < 8; ++e)
      atomicAdd(&sAlpha[mb + e], sigmoidf(q1n + acc[e] + b2n) * w3n);
  }
  __syncthreads();

  // S11: a = sum_l alpha*seq*mask ; a_final = [a|ht]@Wt^T + bt -> ws (bf16)
  if (tid < HDIM) {
    float a = 0.f;
    for (int l = 0; l < SEQL; ++l)
      if (al[l] > 0) a += sAlpha[l] * sHid[l][tid];
    sAvec[tid] = a;
  }
  __syncthreads();
  if (tid < HDIM) {
    float af = bt[tid];
    const float* wr = Wt + tid * (2 * HDIM);
    for (int k = 0; k < HDIM; ++k) af += sAvec[k] * wr[k];
    for (int k = 0; k < HDIM; ++k) af += sHt[k] * wr[HDIM + k];
    a_out[b * HDIM + tid] = (bf16_t)af;
  }
}

// ============================================================================
// Kernel 2: pred = 16 * (a @ emb[1:]^T)   [512, 99999]  (memory-bound)
// One block per 16-column tile of the vocabulary.  The 16x128 f32 emb tile is
// staged into LDS by the Tensor Data Mover (tensor_load_to_lds, 2D D#),
// which also zero-fills out-of-range tail rows per the ISA OOB rule.
// ============================================================================
__global__ __launch_bounds__(256)
void niser_logits_kernel(const bf16_t* __restrict__ a_bf,  // [512][128]
                         const float*  __restrict__ emb,   // [V][128]
                         float*        __restrict__ out) { // [512][V-1]
  __shared__ float sEf[16][HDIM];  // 8 KB: 16 candidate-item embedding rows
  const int tid  = threadIdx.x;
  const int wave = tid >> 5;
  const int lane = tid & 31;
  const int n0   = blockIdx.x * 16;
  const int OC   = VOCAB - 1;

  if (wave == 0) {
    // --- Tensor DMA descriptor (D#), 2D tile: 16 rows x 128 f32 ---
    unsigned lds_off = (unsigned)(uintptr_t)(&sEf[0][0]);
    unsigned long long gaddr =
        (unsigned long long)(uintptr_t)(emb + (size_t)(1 + n0) * HDIM);
    unsigned rows_left = (unsigned)(VOCAB - 1 - n0);  // rows valid from tile start

    su4 g0;
    g0[0] = 1u;                                        // count=1, user mode
    g0[1] = lds_off;                                   // lds_addr
    g0[2] = (unsigned)(gaddr & 0xffffffffu);           // global_addr[31:0]
    g0[3] = (unsigned)((gaddr >> 32) & 0x01ffffffu)    // global_addr[56:32]
            | (2u << 30);                              // type=2 (image)
    su8 g1;
    g1[0] = (2u << 16);                                // data_size=2 -> 4-byte elems
    g1[1] = (128u << 16);                              // tensor_dim0 = 128
    g1[2] = ((rows_left & 0xffffu) << 16);             // tensor_dim1 lo16
    g1[3] = ((rows_left >> 16) & 0xffffu)              // tensor_dim1 hi16
            | (128u << 16);                            // tile_dim0 = 128
    g1[4] = 16u;                                       // tile_dim1=16, tile_dim2=0
    g1[5] = 128u;                                      // tensor_dim0_stride = 128
    g1[6] = 0u;
    g1[7] = 0u;

    asm volatile("tensor_load_to_lds %0, %1" :: "s"(g0), "s"(g1) : "memory");
    __builtin_amdgcn_s_wait_tensorcnt(0);
  }
  __syncthreads();

  for (int mt = 0; mt < 4; ++mt) {
    int m0 = wave * 64 + mt * 16;
    v8f acc = {0.f, 0.f, 0.f, 0.f, 0.f, 0.f, 0.f, 0.f};
    for (int kc = 0; kc < 4; ++kc)
      acc = wmma_bf16(ldA_bf16(a_bf, HDIM, m0, kc, lane),
                      ldB_rowT_f32(&sEf[0][0], HDIM, 0, kc, lane), acc);
    int n = n0 + (lane & 15);
    if (n < OC) {
      int mb = m0 + ((lane & 16) ? 8 : 0);
#pragma unroll
      for (int e = 0; e < 8; ++e)
        out[(size_t)(mb + e) * OC + n] = 16.f * acc[e];
    }
  }
}

// ============================================================================
extern "C" void kernel_launch(void* const* d_in, const int* in_sizes, int n_in,
                              void* d_out, int out_size, void* d_ws, size_t ws_size,
                              hipStream_t stream) {
  const int*   items      = (const int*)  d_in[0];
  const float* Amat       = (const float*)d_in[1];
  const int*   alias_in   = (const int*)  d_in[2];
  const float* emb        = (const float*)d_in[3];
  const float* pos        = (const float*)d_in[4];
  const float* linear_pos = (const float*)d_in[5];
  const float* w_ih       = (const float*)d_in[6];
  const float* w_hh       = (const float*)d_in[7];
  const float* b_ih       = (const float*)d_in[8];
  const float* b_hh       = (const float*)d_in[9];
  const float* b_iah      = (const float*)d_in[10];
  const float* b_oah      = (const float*)d_in[11];
  const float* W_in       = (const float*)d_in[12];
  const float* b_in       = (const float*)d_in[13];
  const float* W_out      = (const float*)d_in[14];
  const float* b_out      = (const float*)d_in[15];
  const float* W1         = (const float*)d_in[16];
  const float* b1         = (const float*)d_in[17];
  const float* W2         = (const float*)d_in[18];
  const float* b2         = (const float*)d_in[19];
  const float* W3         = (const float*)d_in[20];
  const float* Wt         = (const float*)d_in[21];
  const float* bt         = (const float*)d_in[22];

  bf16_t* a_bf = (bf16_t*)d_ws;  // [BATCH][HDIM] bf16 = 128 KB

  niser_session_kernel<<<BATCH, 256, 0, stream>>>(
      items, Amat, alias_in, emb, pos, linear_pos, w_ih, w_hh, b_ih, b_hh,
      b_iah, b_oah, W_in, b_in, W_out, b_out, W1, b1, W2, b2, W3, Wt, bt, a_bf);

  const int OC = VOCAB - 1;
  int nblocks = (OC + 15) / 16;  // 6250
  niser_logits_kernel<<<nblocks, 256, 0, stream>>>(a_bf, emb, (float*)d_out);
}
